// ValueFunction_58969900974226
// MI455X (gfx1250) — matrix operand order repaired and checked by reading, not compiled, not run on hardware
//
#include <hip/hip_runtime.h>
#include <hip/hip_bf16.h>
#include <math.h>

// ---------------------------------------------------------------------------
// Problem constants
// ---------------------------------------------------------------------------
#define NN   163840      // nodes
#define EE   2621440     // edges
#define DIN  10
#define DHID 64
#define DM   74          // d_model
#define DPAD 80          // padded d_model stride (f32 buffers)
#define S_G  320         // nodes per graph
#define B_G  512         // graphs
#define HD   37          // head dim
#define NHEAD 2
#define QKV_LD 224       // padded 3*74 stride (bf16 buffer)

typedef __attribute__((ext_vector_type(16))) __bf16 bf16x16;
typedef __attribute__((ext_vector_type(8)))  float  f32x8;

__device__ __forceinline__ f32x8 wmma_bf16(bf16x16 a, bf16x16 b, f32x8 c) {
  return __builtin_amdgcn_wmma_f32_16x16x32_bf16(false, a, false, b, (short)0, c,
                                                 false, false);
}

// ---------------------------------------------------------------------------
// Workspace layout (bytes).  Sage scratch aliases transformer scratch (dead
// before the transformer starts).  Total ~273 MB.
// ---------------------------------------------------------------------------
constexpr size_t kN = NN, kE = EE;
constexpr size_t OFF_QKV  = 0;                         // bf16 [N,224]
constexpr size_t SZ_QKV   = kN * QKV_LD * 2;
constexpr size_t OFF_ATTN = OFF_QKV + SZ_QKV;          // f32 [N,80]  (alias: h1 [N,64])
constexpr size_t SZ_ATTN  = kN * DPAD * 4;
constexpr size_t OFF_FFH  = OFF_ATTN + SZ_ATTN;        // f32 [N,64]
constexpr size_t SZ_FFH   = kN * DHID * 4;
constexpr size_t OFF_TMP  = OFF_FFH + SZ_FFH;          // f32 [N,80]
constexpr size_t SZ_TMP   = kN * DPAD * 4;
constexpr size_t OFF_EMB  = OFF_TMP + SZ_TMP;          // f32 [N,80] persistent
constexpr size_t SZ_EMB   = kN * DPAD * 4;
// Sage-phase aliases (inside the QKV region):
constexpr size_t OFF_AGG1 = 0;                         // f32 [N,10]
constexpr size_t OFF_CNT  = OFF_AGG1 + kN * DIN * 4;   // f32 [N]
constexpr size_t OFF_AGG2 = OFF_CNT + kN * 4;          // f32 [N,64]
constexpr size_t SAGE_ZERO_BYTES = OFF_AGG2 + kN * DHID * 4;
static_assert(SAGE_ZERO_BYTES <= SZ_QKV, "sage scratch must fit in qkv region");
static_assert(kN * DHID * 4 <= SZ_ATTN, "h1 must fit in attn region");

// ---------------------------------------------------------------------------
// Graph scatter phase
// ---------------------------------------------------------------------------
__global__ void degree_kernel(const long long* __restrict__ ei,
                              float* __restrict__ cnt) {
  size_t e = (size_t)blockIdx.x * blockDim.x + threadIdx.x;
  if (e >= kE) return;
  int dst = (int)ei[kE + e];
  atomicAdd(&cnt[dst], 1.0f);
}

__global__ void scatter1_kernel(const long long* __restrict__ ei,
                                const float* __restrict__ x,
                                float* __restrict__ agg1) {
  size_t e = (size_t)blockIdx.x * blockDim.x + threadIdx.x;
  if (e >= kE) return;
  int src = (int)ei[e];
  int dst = (int)ei[kE + e];
  const float* xs = x + (size_t)src * DIN;
  float* ad = agg1 + (size_t)dst * DIN;
#pragma unroll
  for (int i = 0; i < DIN; ++i) atomicAdd(&ad[i], xs[i]);
}

__global__ void sage1_kernel(const float* __restrict__ x,
                             const float* __restrict__ agg1,
                             const float* __restrict__ cnt,
                             const float* __restrict__ lw,
                             const float* __restrict__ lb,
                             const float* __restrict__ rw,
                             float* __restrict__ h1) {
  size_t t = (size_t)blockIdx.x * blockDim.x + threadIdx.x;
  if (t >= kN * (size_t)DHID) return;
  int o = (int)(t & 63);
  size_t n = t >> 6;
  float invc = 1.0f / fmaxf(cnt[n], 1.0f);
  float acc = lb[o];
#pragma unroll
  for (int i = 0; i < DIN; ++i) {
    acc += agg1[n * DIN + i] * invc * lw[o * DIN + i] + x[n * DIN + i] * rw[o * DIN + i];
  }
  h1[n * DHID + o] = fmaxf(acc, 0.0f);
}

__global__ void scatter2_kernel(const long long* __restrict__ ei,
                                const float* __restrict__ h1,
                                float* __restrict__ agg2) {
  size_t t = (size_t)blockIdx.x * blockDim.x + threadIdx.x;
  if (t >= kE * (size_t)DHID) return;
  size_t e = t >> 6;
  int d = (int)(t & 63);
  int src = (int)ei[e];
  int dst = (int)ei[kE + e];
  atomicAdd(&agg2[(size_t)dst * DHID + d], h1[(size_t)src * DHID + d]);
}

// SAGE layer 2 via WMMA: emb[:,0:64] = relu(mean2 @ lw^T + lb + h1 @ rw^T)
__global__ void __launch_bounds__(256) sage2_kernel(
    const float* __restrict__ agg2, const float* __restrict__ cnt,
    const float* __restrict__ h1, const float* __restrict__ lw,
    const float* __restrict__ lb, const float* __restrict__ rw,
    float* __restrict__ emb) {
  int wave = blockIdx.x * 8 + (threadIdx.x >> 5);
  int mt = wave >> 2, nt = wave & 3;          // 4 N-tiles (64 cols)
  if (mt >= NN / 16) return;
  int lane = threadIdx.x & 31, lm = lane & 15, half = lane >> 4;
  int n0 = nt << 4;
  size_t row = (size_t)(mt << 4) + lm;
  float invc = 1.0f / fmaxf(cnt[row], 1.0f);
  f32x8 c = {};
  {
    float bv = lb[n0 + lm];
#pragma unroll
    for (int r = 0; r < 8; ++r) c[r] = bv;
  }
#pragma unroll
  for (int k0 = 0; k0 < DHID; k0 += 32) {
    bf16x16 a1, a2, b1, b2;
#pragma unroll
    for (int j = 0; j < 16; ++j) {
      int kk = k0 + 8 * half + (j & 7) + ((j >> 3) << 4);
      a1[j] = (__bf16)(agg2[row * DHID + kk] * invc);
      a2[j] = (__bf16)(h1[row * DHID + kk]);
    }
#pragma unroll
    for (int j = 0; j < 16; ++j) {
      int kk = k0 + 8 * half + (j & 7) + ((j >> 3) << 4);
      b1[j] = (__bf16)(lw[(size_t)(n0 + lm) * DHID + kk]);
      b2[j] = (__bf16)(rw[(size_t)(n0 + lm) * DHID + kk]);
    }
    c = wmma_bf16(a1, b1, c);
    c = wmma_bf16(a2, b2, c);
  }
#pragma unroll
  for (int r = 0; r < 8; ++r) {
    size_t erow = (size_t)(mt << 4) + r + 8 * half;
    emb[erow * DPAD + n0 + lm] = fmaxf(c[r], 0.0f);
  }
}

// emb[:,64:74] = x, emb[:,74:80] = 0
__global__ void concat_kernel(const float* __restrict__ x, float* __restrict__ emb) {
  size_t t = (size_t)blockIdx.x * blockDim.x + threadIdx.x;
  if (t >= kN * 16) return;
  size_t n = t >> 4;
  int j = (int)(t & 15);
  emb[n * DPAD + 64 + j] = (j < DIN) ? x[n * DIN + j] : 0.0f;
}

// ---------------------------------------------------------------------------
// Generic WMMA GEMM: C[M,Nout] = A[M,K] @ W[Nout,K]^T + bias  (one wave/tile)
// ---------------------------------------------------------------------------
template <bool RELU, bool OUT_BF16>
__global__ void __launch_bounds__(256) gemm_bias_kernel(
    const float* __restrict__ A, int lda, int M, int K,
    const float* __restrict__ W, int ldw, int Nout,
    const float* __restrict__ bias, void* __restrict__ Cout, int ldc) {
  int wave = blockIdx.x * 8 + (threadIdx.x >> 5);
  int ntn = (Nout + 15) >> 4;
  int mt = wave / ntn;
  int nt = wave - mt * ntn;
  if (mt >= ((M + 15) >> 4)) return;
  int lane = threadIdx.x & 31, lm = lane & 15, half = lane >> 4;
  int m0 = mt << 4, n0 = nt << 4;
  int arow = m0 + lm, wcol = n0 + lm;

  f32x8 c = {};
  {
    float bv = (wcol < Nout) ? bias[wcol] : 0.0f;
#pragma unroll
    for (int r = 0; r < 8; ++r) c[r] = bv;
  }
  for (int k0 = 0; k0 < K; k0 += 32) {
    bf16x16 a, b;
#pragma unroll
    for (int j = 0; j < 16; ++j) {
      int kk = k0 + 8 * half + (j & 7) + ((j >> 3) << 4);
      float v = 0.0f;
      if (arow < M && kk < K) v = A[(size_t)arow * lda + kk];
      a[j] = (__bf16)v;
    }
#pragma unroll
    for (int j = 0; j < 16; ++j) {
      int kk = k0 + 8 * half + (j & 7) + ((j >> 3) << 4);
      float v = 0.0f;
      if (wcol < Nout && kk < K) v = W[(size_t)wcol * ldw + kk];
      b[j] = (__bf16)v;
    }
    c = wmma_bf16(a, b, c);
  }
#pragma unroll
  for (int r = 0; r < 8; ++r) {
    int row = m0 + r + 8 * half;
    int col = n0 + lm;
    if (row < M && col < Nout) {
      float v = c[r];
      if (RELU) v = fmaxf(v, 0.0f);
      if (OUT_BF16)
        ((__bf16*)Cout)[(size_t)row * ldc + col] = (__bf16)v;
      else
        ((float*)Cout)[(size_t)row * ldc + col] = v;
    }
  }
}

// ---------------------------------------------------------------------------
// Attention: one workgroup per (graph, head); 4 waves, K/V staged in LDS.
// qkv (bf16, [N,224]) cols: q 0..73, k 74..147, v 148..221 (per-head 37 each).
// ---------------------------------------------------------------------------
__global__ void __launch_bounds__(128) attn_kernel(const __bf16* __restrict__ qkv,
                                                   float* __restrict__ attn_out) {
  extern __shared__ char smem[];
  __bf16* Ksh = (__bf16*)smem;                 // [320][64]  (d padded to 64)
  __bf16* Vsh = Ksh + S_G * 64;                // [320][48]  (d padded to 48)
  float* Pbase = (float*)(Vsh + S_G * 48);     // 4 waves x [16][320]

  int b = blockIdx.x >> 1;
  int head = blockIdx.x & 1;
  size_t node0 = (size_t)b * S_G;
  int tid = threadIdx.x, wave = tid >> 5, lane = tid & 31;
  int lm = lane & 15, half = lane >> 4;
  float* Pw = Pbase + (size_t)wave * 16 * S_G;

  for (int i = tid; i < S_G * 64; i += 128) {
    int s = i >> 6, d = i & 63;
    float v = (d < HD) ? (float)qkv[(node0 + s) * QKV_LD + DM + head * HD + d] : 0.0f;
    Ksh[i] = (__bf16)v;
  }
  for (int i = tid; i < S_G * 48; i += 128) {
    int s = i / 48, d = i - s * 48;
    float v = (d < HD) ? (float)qkv[(node0 + s) * QKV_LD + 2 * DM + head * HD + d] : 0.0f;
    Vsh[i] = (__bf16)v;
  }
  __syncthreads();

  const float qscale = rsqrtf((float)HD);
  for (int qt = wave; qt < S_G / 16; qt += 4) {
    int q0 = qt * 16;
    // Q fragments (d padded to 64 -> 2 k-steps)
    bf16x16 aq[2];
#pragma unroll
    for (int t = 0; t < 2; ++t) {
#pragma unroll
      for (int j = 0; j < 16; ++j) {
        int d = t * 32 + 8 * half + (j & 7) + ((j >> 3) << 4);
        float v = (d < HD)
                      ? (float)qkv[(node0 + q0 + lm) * QKV_LD + head * HD + d] * qscale
                      : 0.0f;
        aq[t][j] = (__bf16)v;
      }
    }
    // scores S[16][320] = Q @ K^T
    for (int kt = 0; kt < S_G / 16; ++kt) {
      f32x8 c = {};
#pragma unroll
      for (int t = 0; t < 2; ++t) {
        bf16x16 bk;
#pragma unroll
        for (int j = 0; j < 16; ++j) {
          int kd = t * 32 + 8 * half + (j & 7) + ((j >> 3) << 4);
          bk[j] = Ksh[(kt * 16 + lm) * 64 + kd];
        }
        c = wmma_bf16(aq[t], bk, c);
      }
#pragma unroll
      for (int r = 0; r < 8; ++r) Pw[(r + 8 * half) * S_G + kt * 16 + lm] = c[r];
    }
    asm volatile("s_wait_dscnt 0" ::: "memory");
    // softmax: lane handles row lm, column half `half` (0..159 / 160..319)
    {
      int c0 = half * 160, c1 = c0 + 160;
      float mx = -1e30f;
      for (int cdx = c0; cdx < c1; ++cdx) mx = fmaxf(mx, Pw[lm * S_G + cdx]);
      mx = fmaxf(mx, __shfl_xor(mx, 16, 32));
      float sum = 0.0f;
      for (int cdx = c0; cdx < c1; ++cdx) {
        float e = __expf(Pw[lm * S_G + cdx] - mx);
        Pw[lm * S_G + cdx] = e;
        sum += e;
      }
      sum += __shfl_xor(sum, 16, 32);
      float inv = 1.0f / sum;
      for (int cdx = c0; cdx < c1; ++cdx) Pw[lm * S_G + cdx] *= inv;
    }
    asm volatile("s_wait_dscnt 0" ::: "memory");
    // O[16][48] = P @ V
    f32x8 co[3];
#pragma unroll
    for (int nt = 0; nt < 3; ++nt) {
      f32x8 z = {};
      co[nt] = z;
    }
    for (int k0 = 0; k0 < S_G; k0 += 32) {
      bf16x16 ap;
#pragma unroll
      for (int j = 0; j < 16; ++j) {
        int kk = k0 + 8 * half + (j & 7) + ((j >> 3) << 4);
        ap[j] = (__bf16)Pw[lm * S_G + kk];
      }
#pragma unroll
      for (int nt = 0; nt < 3; ++nt) {
        bf16x16 bv;
#pragma unroll
        for (int j = 0; j < 16; ++j) {
          int kk = k0 + 8 * half + (j & 7) + ((j >> 3) << 4);
          bv[j] = Vsh[kk * 48 + nt * 16 + lm];
        }
        co[nt] = wmma_bf16(ap, bv, co[nt]);
      }
    }
#pragma unroll
    for (int nt = 0; nt < 3; ++nt) {
#pragma unroll
      for (int r = 0; r < 8; ++r) {
        int col = nt * 16 + lm;
        if (col < HD)
          attn_out[(node0 + q0 + r + 8 * half) * DPAD + head * HD + col] = co[nt][r];
      }
    }
  }
}

// emb = LayerNorm(emb + delta) (cols 0..73; 74..79 stay zero)
__global__ void resid_ln_kernel(const float* __restrict__ delta,
                                const float* __restrict__ g,
                                const float* __restrict__ beta,
                                float* __restrict__ emb) {
  size_t n = (size_t)blockIdx.x * blockDim.x + threadIdx.x;
  if (n >= kN) return;
  float v[DM];
  float mu = 0.0f;
#pragma unroll
  for (int c = 0; c < DM; ++c) {
    v[c] = emb[n * DPAD + c] + delta[n * DPAD + c];
    mu += v[c];
  }
  mu *= (1.0f / DM);
  float var = 0.0f;
#pragma unroll
  for (int c = 0; c < DM; ++c) {
    float d = v[c] - mu;
    var += d * d;
  }
  var *= (1.0f / DM);
  float inv = rsqrtf(var + 1e-5f);
#pragma unroll
  for (int c = 0; c < DM; ++c) emb[n * DPAD + c] = (v[c] - mu) * inv * g[c] + beta[c];
}

// mean-pool over S then critic MLP; one workgroup per graph
__global__ void __launch_bounds__(128) pool_critic_kernel(
    const float* __restrict__ emb, const float* __restrict__ w1,
    const float* __restrict__ bias1, const float* __restrict__ w2,
    const float* __restrict__ bias2, float* __restrict__ out) {
  __shared__ float gi[DM];
  __shared__ float hh[64];
  int g = blockIdx.x, t = threadIdx.x;
  for (int c = t; c < DM; c += 128) {
    float s = 0.0f;
    for (int r = 0; r < S_G; ++r) s += emb[((size_t)g * S_G + r) * DPAD + c];
    gi[c] = s * (1.0f / S_G);
  }
  __syncthreads();
  if (t < 64) {
    float a = bias1[t];
    for (int c = 0; c < DM; ++c) a += gi[c] * w1[t * DM + c];
    hh[t] = a / (1.0f + __expf(-a));  // SiLU
  }
  __syncthreads();
  if (t == 0) {
    float a = bias2[0];
    for (int o = 0; o < 64; ++o) a += hh[o] * w2[o];
    out[g] = a;
  }
}

// ---------------------------------------------------------------------------
// Host launcher
// ---------------------------------------------------------------------------
static inline unsigned gemm_blocks(int M, int Nout) {
  long tiles = (long)((M + 15) / 16) * ((Nout + 15) / 16);
  return (unsigned)((tiles + 7) / 8);
}

extern "C" void kernel_launch(void* const* d_in, const int* in_sizes, int n_in,
                              void* d_out, int out_size, void* d_ws, size_t ws_size,
                              hipStream_t stream) {
  const float* x = (const float*)d_in[0];
  const long long* ei = (const long long*)d_in[1];
  // d_in[2] = edge_attr (unused by the reference)
  const float* s1_lw = (const float*)d_in[3];
  const float* s1_lb = (const float*)d_in[4];
  const float* s1_rw = (const float*)d_in[5];
  const float* s2_lw = (const float*)d_in[6];
  const float* s2_lb = (const float*)d_in[7];
  const float* s2_rw = (const float*)d_in[8];
  const float* t_in_w = (const float*)d_in[9];
  const float* t_in_b = (const float*)d_in[10];
  const float* t_out_w = (const float*)d_in[11];
  const float* t_out_b = (const float*)d_in[12];
  const float* t_ln1_g = (const float*)d_in[13];
  const float* t_ln1_b = (const float*)d_in[14];
  const float* t_ff1_w = (const float*)d_in[15];
  const float* t_ff1_b = (const float*)d_in[16];
  const float* t_ff2_w = (const float*)d_in[17];
  const float* t_ff2_b = (const float*)d_in[18];
  const float* t_ln2_g = (const float*)d_in[19];
  const float* t_ln2_b = (const float*)d_in[20];
  const float* c_w1 = (const float*)d_in[21];
  const float* c_b1 = (const float*)d_in[22];
  const float* c_w2 = (const float*)d_in[23];
  const float* c_b2 = (const float*)d_in[24];
  float* out = (float*)d_out;

  char* ws = (char*)d_ws;
  float* agg1 = (float*)(ws + OFF_AGG1);
  float* cnt  = (float*)(ws + OFF_CNT);
  float* agg2 = (float*)(ws + OFF_AGG2);
  float* h1   = (float*)(ws + OFF_ATTN);  // alias, dead before attention
  __bf16* qkv = (__bf16*)(ws + OFF_QKV);
  float* attn = (float*)(ws + OFF_ATTN);
  float* ffh  = (float*)(ws + OFF_FFH);
  float* tmp  = (float*)(ws + OFF_TMP);
  float* emb  = (float*)(ws + OFF_EMB);

  // ---- SAGE phase ----
  hipMemsetAsync(ws, 0, SAGE_ZERO_BYTES, stream);
  unsigned eb = (unsigned)((kE + 255) / 256);
  degree_kernel<<<eb, 256, 0, stream>>>(ei, cnt);
  scatter1_kernel<<<eb, 256, 0, stream>>>(ei, x, agg1);
  sage1_kernel<<<(unsigned)(kN * DHID / 256), 256, 0, stream>>>(x, agg1, cnt, s1_lw,
                                                                s1_lb, s1_rw, h1);
  scatter2_kernel<<<(unsigned)(kE * DHID / 256), 256, 0, stream>>>(ei, h1, agg2);
  sage2_kernel<<<(unsigned)(kN / 16 * 4 / 8), 256, 0, stream>>>(agg2, cnt, h1, s2_lw,
                                                                s2_lb, s2_rw, emb);
  concat_kernel<<<(unsigned)(kN * 16 / 256), 256, 0, stream>>>(x, emb);

  // ---- Transformer phase ----
  const int ATTN_LDS = S_G * 64 * 2 + S_G * 48 * 2 + 4 * 16 * S_G * 4;  // 153600 B
  (void)hipFuncSetAttribute(reinterpret_cast<const void*>(attn_kernel),
                            hipFuncAttributeMaxDynamicSharedMemorySize, ATTN_LDS);
  unsigned lnb = (unsigned)((kN + 255) / 256);
  for (int l = 0; l < 2; ++l) {
    // QKV projection -> bf16 [N,224]
    gemm_bias_kernel<false, true><<<gemm_blocks(NN, 3 * DM), 256, 0, stream>>>(
        emb, DPAD, NN, DM, t_in_w + (size_t)l * 3 * DM * DM, DM, 3 * DM,
        t_in_b + (size_t)l * 3 * DM, qkv, QKV_LD);
    // attention
    attn_kernel<<<B_G * NHEAD, 128, ATTN_LDS, stream>>>(qkv, attn);
    // out projection -> tmp
    gemm_bias_kernel<false, false><<<gemm_blocks(NN, DM), 256, 0, stream>>>(
        attn, DPAD, NN, DM, t_out_w + (size_t)l * DM * DM, DM, DM,
        t_out_b + (size_t)l * DM, tmp, DPAD);
    // emb = LN(emb + tmp)
    resid_ln_kernel<<<lnb, 256, 0, stream>>>(tmp, t_ln1_g + (size_t)l * DM,
                                             t_ln1_b + (size_t)l * DM, emb);
    // FF1 (relu) -> ffh
    gemm_bias_kernel<true, false><<<gemm_blocks(NN, DHID), 256, 0, stream>>>(
        emb, DPAD, NN, DM, t_ff1_w + (size_t)l * DHID * DM, DM, DHID,
        t_ff1_b + (size_t)l * DHID, ffh, DHID);
    // FF2 -> tmp
    gemm_bias_kernel<false, false><<<gemm_blocks(NN, DM), 256, 0, stream>>>(
        ffh, DHID, NN, DHID, t_ff2_w + (size_t)l * DM * DHID, DHID, DM,
        t_ff2_b + (size_t)l * DM, tmp, DPAD);
    // emb = LN(emb + tmp)
    resid_ln_kernel<<<lnb, 256, 0, stream>>>(tmp, t_ln2_g + (size_t)l * DM,
                                             t_ln2_b + (size_t)l * DM, emb);
  }

  // ---- Pool + critic ----
  pool_critic_kernel<<<B_G, 128, 0, stream>>>(emb, c_w1, c_b1, c_w2, c_b2, out);
}
